// Mamba_3410204033868
// MI455X (gfx1250) — matrix operand order repaired and checked by reading, not compiled, not run on hardware
//
#include <hip/hip_runtime.h>
#include <hip/hip_bf16.h>

// ---------------- Problem constants (from reference) ----------------
#define B_    2
#define L_    1024
#define DM    768          // d_model
#define NL    4            // n_layer
#define DI    1536         // d_inner
#define DS    16           // d_state
#define DC    4            // d_conv
#define DTR   48           // dt_rank
#define DTP   64           // dt_rank zero-padded to a multiple of 32 for WMMA
#define XPN   (DTR + 2*DS) // 80, x_proj output width
#define ROWS  (B_*L_)      // 2048 token rows

typedef __bf16 bf16_t;
typedef bf16_t v16bf __attribute__((ext_vector_type(16)));
typedef bf16_t v8bf  __attribute__((ext_vector_type(8)));
typedef float  v8f   __attribute__((ext_vector_type(8)));

// ---------------- Elementwise helpers ----------------

__global__ void cvt_f32_bf16_kernel(const float* __restrict__ src,
                                    bf16_t* __restrict__ dst, int n) {
  int i = blockIdx.x * blockDim.x + threadIdx.x;
  if (i < n) dst[i] = (bf16_t)src[i];
}

// Convert rows x srcK f32 -> rows x dstK bf16, zero-padding cols >= srcK
__global__ void cvt_pad_bf16_kernel(const float* __restrict__ src,
                                    bf16_t* __restrict__ dst,
                                    int rows, int srcK, int dstK) {
  int i = blockIdx.x * blockDim.x + threadIdx.x;
  if (i >= rows * dstK) return;
  int r = i / dstK, c = i % dstK;
  dst[i] = (c < srcK) ? (bf16_t)src[(size_t)r * srcK + c] : (bf16_t)0.0f;
}

__global__ void embed_kernel(const int* __restrict__ seq,
                             const float* __restrict__ emb,
                             float* __restrict__ residual) {
  int i = blockIdx.x * blockDim.x + threadIdx.x;   // over ROWS*DM
  if (i >= ROWS * DM) return;
  int r = i / DM, c = i % DM;
  residual[i] = emb[(size_t)seq[r] * DM + c];
}

// RMSNorm over DM, one block per row, bf16 output (feeds WMMA A-matrix)
__global__ void rmsnorm_bf16_kernel(const float* __restrict__ x,
                                    const float* __restrict__ w,
                                    bf16_t* __restrict__ out) {
  int row = blockIdx.x;
  const float* xr = x + (size_t)row * DM;
  __shared__ float red[256];
  int t = threadIdx.x;
  float ss = 0.f;
  for (int i = t; i < DM; i += 256) { float v = xr[i]; ss += v * v; }
  red[t] = ss; __syncthreads();
  for (int st = 128; st > 0; st >>= 1) { if (t < st) red[t] += red[t + st]; __syncthreads(); }
  float rinv = rsqrtf(red[0] / (float)DM + 1e-5f);
  for (int i = t; i < DM; i += 256)
    out[(size_t)row * DM + i] = (bf16_t)(xr[i] * rinv * w[i]);
}

// Depthwise causal conv (k=4) + bias + SiLU over x half of xz; dual f32/bf16 out
__global__ void conv_silu_kernel(const float* __restrict__ xz,
                                 const float* __restrict__ cw,
                                 const float* __restrict__ cb,
                                 float* __restrict__ xc,
                                 bf16_t* __restrict__ xcb) {
  int i = blockIdx.x * blockDim.x + threadIdx.x;   // over ROWS*DI
  if (i >= ROWS * DI) return;
  int d = i % DI;
  int l = (i / DI) % L_;
  int b = i / (DI * L_);
  float acc = cb[d];
  #pragma unroll
  for (int k = 0; k < DC; k++) {
    int ls = l + k - (DC - 1);
    if (ls >= 0)
      acc += xz[((size_t)(b * L_ + ls)) * (2 * DI) + d] * cw[d * DC + k];
  }
  float s = acc / (1.f + __expf(-acc));            // SiLU
  xc[i]  = s;
  xcb[i] = (bf16_t)s;
}

// Slice dt (first DTR cols of xdbl) into bf16, zero-padded to DTP cols
__global__ void slice_dt_kernel(const float* __restrict__ xdbl,
                                bf16_t* __restrict__ dtbf) {
  int i = blockIdx.x * blockDim.x + threadIdx.x;   // over ROWS*DTP
  if (i >= ROWS * DTP) return;
  int r = i / DTP, c = i % DTP;
  dtbf[i] = (c < DTR) ? (bf16_t)xdbl[(size_t)r * XPN + c] : (bf16_t)0.0f;
}

// In-place softplus(x + dt_b[d])
__global__ void softplus_kernel(float* __restrict__ delta,
                                const float* __restrict__ dt_b) {
  int i = blockIdx.x * blockDim.x + threadIdx.x;   // over ROWS*DI
  if (i >= ROWS * DI) return;
  float v = delta[i] + dt_b[i % DI];
  delta[i] = (v > 20.f) ? v : log1pf(__expf(v));
}

// Selective scan: one thread per (b,d) channel, 16-wide state in registers,
// fuses D-skip and SiLU(z) gate; emits bf16 for the out-proj GEMM.
__global__ void scan_kernel(const float* __restrict__ delta,
                            const float* __restrict__ xconv,
                            const float* __restrict__ xdbl,
                            const float* __restrict__ xz,
                            const float* __restrict__ A_log,
                            const float* __restrict__ D_skip,
                            bf16_t* __restrict__ ybf) {
  int tid = blockIdx.x * blockDim.x + threadIdx.x;
  if (tid >= B_ * DI) return;
  int b = tid / DI, d = tid % DI;
  float A_[DS], h[DS];
  #pragma unroll
  for (int s = 0; s < DS; s++) { A_[s] = -__expf(A_log[d * DS + s]); h[s] = 0.f; }
  float dsk = D_skip[d];
  for (int l = 0; l < L_; l++) {
    size_t row = (size_t)b * L_ + l;
    float dt = delta[row * DI + d];
    float u  = xconv[row * DI + d];
    float du = dt * u;
    const float4* bc = (const float4*)(xdbl + row * XPN + DTR); // 16B aligned
    float4 bv[4], cv[4];
    #pragma unroll
    for (int q = 0; q < 4; q++) { bv[q] = bc[q]; cv[q] = bc[4 + q]; }
    const float* bs = (const float*)bv;
    const float* cs = (const float*)cv;
    float y = 0.f;
    #pragma unroll
    for (int s = 0; s < DS; s++) {
      float dA = __expf(dt * A_[s]);
      h[s] = dA * h[s] + du * bs[s];
      y += h[s] * cs[s];
    }
    float zv = xz[row * (2 * DI) + DI + d];
    float sz = zv / (1.f + __expf(-zv));
    ybf[row * DI + d] = (bf16_t)((y + u * dsk) * sz);
  }
}

// ---------------- WMMA GEMM: Y[M,N] (+=) = Xbf16[M,K] . Wbf16[N,K]^T --------
// Preconditions: M%16==0, N%(16*NT)==0, K%32==0 (enforced by buffer padding).
// Each wave computes a 16 x (16*NT) output strip. Explicit double buffering:
// fragments for K-step kb+32 are loaded into `nxt` while the WMMAs consume
// `cur` (loaded the previous iteration). The loop-carried dependence forces
// distinct register live ranges, so the WMMA group never waits on in-flight
// loads. Final iteration clamps the prefetch K-offset to 0 (uniform select,
// no divergence, no OOB); that load set is discarded. EXEC is all-ones at the
// WMMAs (no non-uniform control flow).
template <int NT> struct Frags { v16bf a; v16bf b[NT]; };

template <int NT>
__device__ __forceinline__ void load_frags(const bf16_t* __restrict__ xr,
                                           const bf16_t* const* wr,
                                           int kb, Frags<NT>& f) {
  v8bf a0 = *(const v8bf*)(xr + kb);
  v8bf a1 = *(const v8bf*)(xr + kb + 16);
  v8bf w0[NT], w1[NT];
  #pragma unroll
  for (int t = 0; t < NT; t++) {
    w0[t] = *(const v8bf*)(wr[t] + kb);
    w1[t] = *(const v8bf*)(wr[t] + kb + 16);
  }
  #pragma unroll
  for (int i = 0; i < 8; i++) { f.a[i] = a0[i]; f.a[8 + i] = a1[i]; }
  #pragma unroll
  for (int t = 0; t < NT; t++)
    #pragma unroll
    for (int i = 0; i < 8; i++) { f.b[t][i] = w0[t][i]; f.b[t][8 + i] = w1[t][i]; }
}

template <int NT, int ADD>
__global__ __launch_bounds__(256)
void wmma_gemm_kernel(const bf16_t* __restrict__ X, const bf16_t* __restrict__ W,
                      float* __restrict__ Y, int M, int N, int K) {
  int lane = threadIdx.x & 31;
  int wid  = threadIdx.x >> 5;
  int groupsN = N / (16 * NT);
  int group = blockIdx.x * (blockDim.x >> 5) + wid;
  if (group >= (M >> 4) * groupsN) return;          // uniform per wave
  int m0 = (group / groupsN) << 4;
  int n0 = (group % groupsN) * (16 * NT);
  int lh  = lane >> 4;                              // lane half (0/1)
  int l15 = lane & 15;
  // A 16x32 fragment per lane: row m0+l15, two contiguous 8-elem runs at
  // K-offsets lh*8 and lh*8+16 (ISA A-matrix 16-bit layout).
  const bf16_t* xr = X + (size_t)(m0 + l15) * K + lh * 8;
  // B 32x16 fragment per lane: W row n, two contiguous 8-elem runs at
  // K-offsets lh*8 and lh*8+16 (matches A).
  const bf16_t* wr[NT];
  #pragma unroll
  for (int t = 0; t < NT; t++)
    wr[t] = W + (size_t)(n0 + t * 16 + l15) * K + lh * 8;

  v8f zero = {};
  v8f acc[NT];
  #pragma unroll
  for (int t = 0; t < NT; t++) acc[t] = zero;

  Frags<NT> cur, nxt;
  load_frags<NT>(xr, wr, 0, cur);
  #pragma unroll 2
  for (int kb = 0; kb < K; kb += 32) {
    int kn = (kb + 32 < K) ? (kb + 32) : 0;        // uniform scalar select
    load_frags<NT>(xr, wr, kn, nxt);               // in-flight while WMMAs run
    __builtin_prefetch(xr + kb + 64, 0, 1);        // global_prefetch_b8
    #pragma unroll
    for (int t = 0; t < NT; t++)
      acc[t] = __builtin_amdgcn_wmma_f32_16x16x32_bf16(
          /*neg_a=*/false, cur.a, /*neg_b=*/false, cur.b[t],
          /*c_mod=*/(short)0, acc[t], /*reuse_a=*/false, /*reuse_b=*/false);
    cur = nxt;
  }
  // C/D layout: VGPR r -> M = m0 + lh*8 + r, N = n0 + t*16 + (lane&15)
  #pragma unroll
  for (int t = 0; t < NT; t++) {
    #pragma unroll
    for (int r = 0; r < 8; r++) {
      int m = m0 + lh * 8 + r;
      int n = n0 + t * 16 + l15;
      size_t o = (size_t)m * N + n;
      float v = acc[t][r];
      if (ADD) v += Y[o];
      Y[o] = v;
    }
  }
}

// Final RMSNorm of residual[b, last, :] where last = sum(mask[b]) - 1
__global__ void final_norm_kernel(const float* __restrict__ residual,
                                  const int* __restrict__ mask,
                                  const float* __restrict__ normf_w,
                                  float* __restrict__ out) {
  int b = blockIdx.x;
  __shared__ float red[256];
  __shared__ int  ired[256];
  int t = threadIdx.x;
  int s = 0;
  for (int i = t; i < L_; i += 256) s += mask[b * L_ + i];
  ired[t] = s; __syncthreads();
  for (int st = 128; st > 0; st >>= 1) { if (t < st) ired[t] += ired[t + st]; __syncthreads(); }
  int last = ired[0] - 1;
  const float* row = residual + ((size_t)b * L_ + last) * DM;
  float ss = 0.f;
  for (int i = t; i < DM; i += 256) { float v = row[i]; ss += v * v; }
  red[t] = ss; __syncthreads();
  for (int st = 128; st > 0; st >>= 1) { if (t < st) red[t] += red[t + st]; __syncthreads(); }
  float rinv = rsqrtf(red[0] / (float)DM + 1e-5f);
  for (int i = t; i < DM; i += 256) out[(size_t)b * DM + i] = row[i] * rinv * normf_w[i];
}

// ---------------- Host orchestration ----------------

static inline int eltGrid(long n) { return (int)((n + 255) / 256); }
static inline int gemmGrid(int M, int N, int nt) {
  long groups = (long)(M >> 4) * (N / (16 * nt));
  return (int)((groups + 7) / 8);                   // 8 waves (strips) per block
}

extern "C" void kernel_launch(void* const* d_in, const int* in_sizes, int n_in,
                              void* d_out, int out_size, void* d_ws, size_t ws_size,
                              hipStream_t stream) {
  const int*   seq     = (const int*)  d_in[0];
  const int*   mask    = (const int*)  d_in[1];
  const float* emb     = (const float*)d_in[2];
  const float* norm_w  = (const float*)d_in[3];
  const float* in_w    = (const float*)d_in[4];
  const float* conv_w  = (const float*)d_in[5];
  const float* conv_b  = (const float*)d_in[6];
  const float* xp_w    = (const float*)d_in[7];
  const float* dt_w    = (const float*)d_in[8];
  const float* dt_b    = (const float*)d_in[9];
  const float* A_log   = (const float*)d_in[10];
  const float* D_skip  = (const float*)d_in[11];
  const float* out_w   = (const float*)d_in[12];
  const float* normf_w = (const float*)d_in[13];
  float* out = (float*)d_out;

  // Workspace carving (256B aligned)
  char* ws = (char*)d_ws;
  size_t off = 0;
  auto carve = [&](size_t bytes) -> void* {
    void* p = ws + off;
    off = (off + bytes + 255) & ~(size_t)255;
    return p;
  };
  float*  residual = (float*) carve((size_t)ROWS * DM * 4);
  bf16_t* hnBf     = (bf16_t*)carve((size_t)ROWS * DM * 2);
  float*  xz       = (float*) carve((size_t)ROWS * 2 * DI * 4);
  float*  xcF      = (float*) carve((size_t)ROWS * DI * 4);
  bf16_t* xcBf     = (bf16_t*)carve((size_t)ROWS * DI * 2);
  float*  xdbl     = (float*) carve((size_t)ROWS * XPN * 4);
  bf16_t* dtBf     = (bf16_t*)carve((size_t)ROWS * DTP * 2);
  float*  deltaB   = (float*) carve((size_t)ROWS * DI * 4);
  bf16_t* yBf      = (bf16_t*)carve((size_t)ROWS * DI * 2);
  bf16_t* wInBf    = (bf16_t*)carve((size_t)2 * DI * DM * 2);
  bf16_t* wXpBf    = (bf16_t*)carve((size_t)XPN * DI * 2);
  bf16_t* wDtBf    = (bf16_t*)carve((size_t)DI * DTP * 2);
  bf16_t* wOutBf   = (bf16_t*)carve((size_t)DM * DI * 2);

  // Embedding gather -> residual
  embed_kernel<<<eltGrid((long)ROWS * DM), 256, 0, stream>>>(seq, emb, residual);

  for (int i = 0; i < NL; i++) {
    // Per-layer weight conversion f32 -> bf16 (deterministic, every call)
    { long n = (long)2 * DI * DM;
      cvt_f32_bf16_kernel<<<eltGrid(n), 256, 0, stream>>>(in_w + (size_t)i * n, wInBf, (int)n); }
    { long n = (long)XPN * DI;
      cvt_f32_bf16_kernel<<<eltGrid(n), 256, 0, stream>>>(xp_w + (size_t)i * n, wXpBf, (int)n); }
    { long n = (long)DI * DTP;    // dt_w zero-padded from K=48 to K=64
      cvt_pad_bf16_kernel<<<eltGrid(n), 256, 0, stream>>>(
          dt_w + (size_t)i * DI * DTR, wDtBf, DI, DTR, DTP); }
    { long n = (long)DM * DI;
      cvt_f32_bf16_kernel<<<eltGrid(n), 256, 0, stream>>>(out_w + (size_t)i * n, wOutBf, (int)n); }

    // RMSNorm -> bf16 activations
    rmsnorm_bf16_kernel<<<ROWS, 256, 0, stream>>>(residual, norm_w + (size_t)i * DM, hnBf);

    // xz = hn . in_w^T   [2048 x 3072], K=768
    wmma_gemm_kernel<4, 0><<<gemmGrid(ROWS, 2 * DI, 4), 256, 0, stream>>>(
        hnBf, wInBf, xz, ROWS, 2 * DI, DM);

    // causal conv + SiLU on x half
    conv_silu_kernel<<<eltGrid((long)ROWS * DI), 256, 0, stream>>>(
        xz, conv_w + (size_t)i * DI * DC, conv_b + (size_t)i * DI, xcF, xcBf);

    // xdbl = x . x_proj_w^T   [2048 x 80], K=1536 (N=80 -> 1 tile per wave)
    wmma_gemm_kernel<1, 0><<<gemmGrid(ROWS, XPN, 1), 256, 0, stream>>>(
        xcBf, wXpBf, xdbl, ROWS, XPN, DI);

    // dt slice -> bf16 (padded K), then delta_pre = dt . dt_w^T [2048 x 1536]
    slice_dt_kernel<<<eltGrid((long)ROWS * DTP), 256, 0, stream>>>(xdbl, dtBf);
    wmma_gemm_kernel<4, 0><<<gemmGrid(ROWS, DI, 4), 256, 0, stream>>>(
        dtBf, wDtBf, deltaB, ROWS, DI, DTP);

    // delta = softplus(delta_pre + dt_b)
    softplus_kernel<<<eltGrid((long)ROWS * DI), 256, 0, stream>>>(
        deltaB, dt_b + (size_t)i * DI);

    // selective scan (fuses D-skip + SiLU(z) gate) -> yBf
    scan_kernel<<<(B_ * DI + 255) / 256, 256, 0, stream>>>(
        deltaB, xcF, xdbl, xz,
        A_log + (size_t)i * DI * DS, D_skip + (size_t)i * DI, yBf);

    // residual += y . out_w^T   [2048 x 768], K=1536, fused add in epilogue
    wmma_gemm_kernel<4, 1><<<gemmGrid(ROWS, DM, 4), 256, 0, stream>>>(
        yBf, wOutBf, residual, ROWS, DM, DI);
  }

  // Final norm at last valid position per batch -> d_out [2 x 768]
  final_norm_kernel<<<B_, 256, 0, stream>>>(residual, mask, normf_w, out);

  (void)in_sizes; (void)n_in; (void)out_size; (void)ws_size;
}
